// GraphSAGE_80547816669618
// MI455X (gfx1250) — matrix operand order repaired and checked by reading, not compiled, not run on hardware
//
#include <hip/hip_runtime.h>
#include <math.h>

#define N_NODES 50000
#define N_EDGES 600000
#define DIM     128
#define OUTC    2
#define NLAYERS 12
#define DD      (DIM * DIM)
#define MTILES  (N_NODES / 16)   // 3125, exact

typedef __attribute__((ext_vector_type(16))) __bf16 v16bf;
typedef __attribute__((ext_vector_type(8)))  float  v8f;

// POD 16-byte vector so it can live in a union (HIP uint4 has ctors)
struct __align__(16) U4 { unsigned x, y, z, w; };

union Frag { U4 q[2]; unsigned u[8]; v16bf v; };

// fp32 -> bf16 round-to-nearest-even (single element, epilogue stores)
static __device__ __forceinline__ unsigned short f2bf(float f) {
    union { float f; unsigned u; } v; v.f = f;
    unsigned r = v.u + 0x7FFFu + ((v.u >> 16) & 1u);
    return (unsigned short)(r >> 16);
}
// packed pair: use native v_cvt_pk_bf16_f32 when the toolchain exposes it
#if __has_builtin(__builtin_amdgcn_cvt_pk_bf16_f32)
typedef __attribute__((ext_vector_type(2))) __bf16 v2bf;
static __device__ __forceinline__ unsigned bfpack2(float lo, float hi) {
    union { v2bf v; unsigned u; } c;
    c.v = __builtin_amdgcn_cvt_pk_bf16_f32(lo, hi);
    return c.u;
}
#else
static __device__ __forceinline__ unsigned bfpack2(float lo, float hi) {
    return (unsigned)f2bf(lo) | ((unsigned)f2bf(hi) << 16);
}
#endif
static __device__ __forceinline__ float bf2f(unsigned short b) {
    union { unsigned u; float f; } v; v.u = ((unsigned)b) << 16;
    return v.f;
}
// aggregation finalize fused into A-load: mean scale or max(-inf)->0
template <int MODE>
static __device__ __forceinline__ float4 xform4(float4 f, float s) {
    if (MODE == 2) {
        f.x = (f.x < -3.0e38f) ? 0.0f : f.x;
        f.y = (f.y < -3.0e38f) ? 0.0f : f.y;
        f.z = (f.z < -3.0e38f) ? 0.0f : f.z;
        f.w = (f.w < -3.0e38f) ? 0.0f : f.w;
    } else if (MODE == 1) {
        f.x *= s; f.y *= s; f.z *= s; f.w *= s;
    }
    return f;
}

// ---------------------------------------------------------------------------
// bulk fp32 -> bf16 (x + weights, once per call)
// ---------------------------------------------------------------------------
__global__ void cvt_f32_bf16(const float* __restrict__ in,
                             unsigned short* __restrict__ out, int n) {
    int i = (blockIdx.x * blockDim.x + threadIdx.x) * 2;
    if (i >= n) return;
    float2 p = *(const float2*)(in + i);
    *(unsigned*)(out + i) = bfpack2(p.x, p.y);
}

// ---------------------------------------------------------------------------
// agg init: 0 for sum/mean, -inf for max; zero degree counts
// ---------------------------------------------------------------------------
__global__ void sage_init_agg(float* __restrict__ agg, float* __restrict__ cnt,
                              int mode) {
    int idx = blockIdx.x * blockDim.x + threadIdx.x;
    if (idx < N_NODES * DIM)
        agg[idx] = (mode == 2) ? -__builtin_inff() : 0.0f;
    if (idx < N_NODES) cnt[idx] = 0.0f;
}

// ---------------------------------------------------------------------------
// scatter-reduce: 16 lanes/edge, one b128 gather of 8 bf16 + 8 fp32 atomics
// ---------------------------------------------------------------------------
template <int MODE>
__global__ void sage_scatter(const unsigned short* __restrict__ h,
                             const int* __restrict__ src,
                             const int* __restrict__ dst,
                             float* __restrict__ agg,
                             float* __restrict__ cnt) {
    int idx = blockIdx.x * blockDim.x + threadIdx.x;
    int e = idx >> 4;
    if (e >= N_EDGES) return;
    int f = (idx & 15) * 8;
    int s = src[e], d = dst[e];
    U4 v = *(const U4*)(h + (size_t)s * DIM + f);
    float x0 = bf2f((unsigned short)(v.x & 0xFFFF));
    float x1 = bf2f((unsigned short)(v.x >> 16));
    float x2 = bf2f((unsigned short)(v.y & 0xFFFF));
    float x3 = bf2f((unsigned short)(v.y >> 16));
    float x4 = bf2f((unsigned short)(v.z & 0xFFFF));
    float x5 = bf2f((unsigned short)(v.z >> 16));
    float x6 = bf2f((unsigned short)(v.w & 0xFFFF));
    float x7 = bf2f((unsigned short)(v.w >> 16));
    float* ap = agg + (size_t)d * DIM + f;
    if (MODE == 2) {
        atomicMax(ap + 0, x0); atomicMax(ap + 1, x1);
        atomicMax(ap + 2, x2); atomicMax(ap + 3, x3);
        atomicMax(ap + 4, x4); atomicMax(ap + 5, x5);
        atomicMax(ap + 6, x6); atomicMax(ap + 7, x7);
    } else {
        atomicAdd(ap + 0, x0); atomicAdd(ap + 1, x1);
        atomicAdd(ap + 2, x2); atomicAdd(ap + 3, x3);
        atomicAdd(ap + 4, x4); atomicAdd(ap + 5, x5);
        atomicAdd(ap + 6, x6); atomicAdd(ap + 7, x7);
        if (MODE == 1 && f == 0) atomicAdd(cnt + d, 1.0f);
    }
}

// ---------------------------------------------------------------------------
// fused layer GEMM: out = relu(T(agg)@Wagg^T [+ h@Wroot^T] + bias), bf16 WMMA
// One wave per 16-row strip; 8 f32 accumulators; B fragments double-buffered
// so loads for tile jt+1 are in flight while tile jt's WMMAs execute.
// A layout: lanes 0-15 K in {0..7,16..23}+koff; lanes 16-31 {8..15,24..31}.
// B layout: lane = col j (j=lane&15), K same as A: {0..7,16..23}+8*khalf+koff.
// ---------------------------------------------------------------------------
template <bool HAS_AGG, int MODE>
__global__ __launch_bounds__(128)
void sage_gemm_bf16(const float* __restrict__ Aagg,
                    const float* __restrict__ cnt,
                    const unsigned short* __restrict__ Hroot,   // bf16 [N][128]
                    const unsigned short* __restrict__ WaggB,   // bf16 [128][128]
                    const unsigned short* __restrict__ WrootB,  // bf16 [128][128]
                    const float* __restrict__ bias,
                    unsigned short* __restrict__ outbf) {
    const int lane  = threadIdx.x & 31;
    const int wave  = threadIdx.x >> 5;
    const int mtile = blockIdx.x * 4 + wave;
    if (mtile >= MTILES) return;                 // wave-uniform exit
    const int mbase = mtile * 16;
    const int mrow  = mbase + (lane & 15);
    const int khalf = lane >> 4;
    const int jl    = lane & 15;

    float scal = 1.0f;
    if (HAS_AGG && MODE == 1) scal = 1.0f / fmaxf(cnt[mrow], 1.0f);

    const float*          aRow = Aagg + (size_t)mrow * DIM;
    const unsigned short* hRow = Hroot + (size_t)mrow * DIM;

    v8f zero = {0.f, 0.f, 0.f, 0.f, 0.f, 0.f, 0.f, 0.f};
    v8f acc[8];
#pragma unroll
    for (int jt = 0; jt < 8; ++jt) acc[jt] = zero;

    for (int kb = 0; kb < 4; ++kb) {
        const int koff = kb * 32;

        // root-path A fragment: bf16, two b128 loads, zero conversion
        Frag ar;
        ar.q[0] = *(const U4*)(hRow + koff + 8 * khalf);
        ar.q[1] = *(const U4*)(hRow + koff + 16 + 8 * khalf);

        // agg-path A fragment: fp32 -> finalize -> bf16 pack
        Frag aa;
        if (HAS_AGG) {
            const float4* p0 = (const float4*)(aRow + koff + 8 * khalf);
            const float4* p1 = (const float4*)(aRow + koff + 16 + 8 * khalf);
            float4 f0 = xform4<MODE>(p0[0], scal);
            float4 f1 = xform4<MODE>(p0[1], scal);
            float4 f2 = xform4<MODE>(p1[0], scal);
            float4 f3 = xform4<MODE>(p1[1], scal);
            aa.u[0] = bfpack2(f0.x, f0.y); aa.u[1] = bfpack2(f0.z, f0.w);
            aa.u[2] = bfpack2(f1.x, f1.y); aa.u[3] = bfpack2(f1.z, f1.w);
            aa.u[4] = bfpack2(f2.x, f2.y); aa.u[5] = bfpack2(f2.z, f2.w);
            aa.u[6] = bfpack2(f3.x, f3.y); aa.u[7] = bfpack2(f3.z, f3.w);
        }

        const size_t wbase = (size_t)jl * DIM + koff + 8 * khalf;

        // software-pipelined sweep over the 8 column tiles
        Frag bn[2], br[2];
        if (HAS_AGG) {
            bn[0].q[0] = *(const U4*)(WaggB + wbase);
            bn[0].q[1] = *(const U4*)(WaggB + wbase + 16);
        }
        br[0].q[0] = *(const U4*)(WrootB + wbase);
        br[0].q[1] = *(const U4*)(WrootB + wbase + 16);
#pragma unroll
        for (int jt = 0; jt < 8; ++jt) {
            const int cur = jt & 1, nxt = cur ^ 1;
            if (jt < 7) {
                const size_t wb = wbase + (size_t)(jt + 1) * 16 * DIM;
                if (HAS_AGG) {
                    bn[nxt].q[0] = *(const U4*)(WaggB + wb);
                    bn[nxt].q[1] = *(const U4*)(WaggB + wb + 16);
                }
                br[nxt].q[0] = *(const U4*)(WrootB + wb);
                br[nxt].q[1] = *(const U4*)(WrootB + wb + 16);
            }
            if (HAS_AGG)
                acc[jt] = __builtin_amdgcn_wmma_f32_16x16x32_bf16(
                    false, aa.v, false, bn[cur].v, (short)0, acc[jt], false, false);
            acc[jt] = __builtin_amdgcn_wmma_f32_16x16x32_bf16(
                false, ar.v, false, br[cur].v, (short)0, acc[jt], false, false);
        }
    }

    // epilogue: bias + relu, store next h in bf16
#pragma unroll
    for (int jt = 0; jt < 8; ++jt) {
        const int jcol = jt * 16 + jl;
        const float bj = bias[jcol];
#pragma unroll
        for (int r = 0; r < 8; ++r) {
            // C/D layout: VGPR r -> M = r (lanes 0-15) / r+8 (lanes 16-31)
            const int row = mbase + r + 8 * khalf;
            outbf[(size_t)row * DIM + jcol] = f2bf(fmaxf(acc[jt][r] + bj, 0.0f));
        }
    }
}

// ---------------------------------------------------------------------------
// final 128 -> 2 projection from bf16 h (fp32 VALU; trivial)
// ---------------------------------------------------------------------------
__global__ void sage_out_proj(const unsigned short* __restrict__ h,
                              const float* __restrict__ W2,
                              const float* __restrict__ b2,
                              float* __restrict__ out) {
    int n = blockIdx.x * blockDim.x + threadIdx.x;
    if (n >= N_NODES) return;
    float a0 = b2[0], a1 = b2[1];
    const unsigned short* hp = h + (size_t)n * DIM;
#pragma unroll 4
    for (int k = 0; k < DIM; ++k) {
        float hv = bf2f(hp[k]);
        a0 += hv * W2[k];
        a1 += hv * W2[DIM + k];
    }
    out[(size_t)n * OUTC + 0] = a0;
    out[(size_t)n * OUTC + 1] = a1;
}

extern "C" void kernel_launch(void* const* d_in, const int* in_sizes, int n_in,
                              void* d_out, int out_size, void* d_ws, size_t ws_size,
                              hipStream_t stream) {
    const float* x  = (const float*)d_in[0];
    const int*   ei = (const int*)d_in[1];
    const float* Wl = (const float*)d_in[2];
    const float* bl = (const float*)d_in[3];
    const float* Wr = (const float*)d_in[4];
    const float* W1 = (const float*)d_in[5];
    const float* b1 = (const float*)d_in[6];
    const float* W2 = (const float*)d_in[7];
    const float* b2 = (const float*)d_in[8];
    const int* src = ei;            // edge_index[0]
    const int* dst = ei + N_EDGES;  // edge_index[1]

    // workspace: fp32 agg + cnt first (16B-aligned), then bf16 arrays
    float* agg = (float*)d_ws;                                  // N*DIM f32
    float* cnt = agg + (size_t)N_NODES * DIM;                   // N f32
    unsigned short* hbfA = (unsigned short*)(cnt + N_NODES);    // N*DIM bf16
    unsigned short* hbfB = hbfA + (size_t)N_NODES * DIM;
    unsigned short* WlB  = hbfB + (size_t)N_NODES * DIM;        // 12*DD bf16
    unsigned short* WrB  = WlB + (size_t)NLAYERS * DD;
    unsigned short* W1B  = WrB + (size_t)NLAYERS * DD;

    static const int AGGR[NLAYERS] = {0,1,2, 0,1,2, 0,1,2, 0,2,1};

    const dim3 blk(256);
    // one-time conversions (per call; deterministic)
    cvt_f32_bf16<<<(N_NODES * DIM / 2 + 255) / 256, blk, 0, stream>>>(x, hbfA, N_NODES * DIM);
    cvt_f32_bf16<<<(NLAYERS * DD / 2 + 255) / 256, blk, 0, stream>>>(Wl, WlB, NLAYERS * DD);
    cvt_f32_bf16<<<(NLAYERS * DD / 2 + 255) / 256, blk, 0, stream>>>(Wr, WrB, NLAYERS * DD);
    cvt_f32_bf16<<<(DD / 2 + 255) / 256, blk, 0, stream>>>(W1, W1B, DD);

    const dim3 initGrid((N_NODES * DIM + 255) / 256);
    const dim3 scatGrid((N_EDGES * 16 + 255) / 256);
    const dim3 gemmGrid((MTILES + 3) / 4);      // 4 row-strip waves per block

    unsigned short* cur = hbfA;
    for (int i = 0; i < NLAYERS; ++i) {
        const int mode = AGGR[i];
        sage_init_agg<<<initGrid, blk, 0, stream>>>(agg, cnt, mode);
        unsigned short* nxt = (cur == hbfA) ? hbfB : hbfA;
        const unsigned short* wl = WlB + (size_t)i * DD;
        const unsigned short* wr = WrB + (size_t)i * DD;
        const float*          bi = bl + (size_t)i * DIM;
        if (mode == 0) {
            sage_scatter<0><<<scatGrid, blk, 0, stream>>>(cur, src, dst, agg, cnt);
            sage_gemm_bf16<true, 0><<<gemmGrid, dim3(128), 0, stream>>>(
                agg, cnt, cur, wl, wr, bi, nxt);
        } else if (mode == 1) {
            sage_scatter<1><<<scatGrid, blk, 0, stream>>>(cur, src, dst, agg, cnt);
            sage_gemm_bf16<true, 1><<<gemmGrid, dim3(128), 0, stream>>>(
                agg, cnt, cur, wl, wr, bi, nxt);
        } else {
            sage_scatter<2><<<scatGrid, blk, 0, stream>>>(cur, src, dst, agg, cnt);
            sage_gemm_bf16<true, 2><<<gemmGrid, dim3(128), 0, stream>>>(
                agg, cnt, cur, wl, wr, bi, nxt);
        }
        cur = nxt;
    }
    // h = relu(h @ W1^T + b1): root path only (Aagg path compiled out)
    unsigned short* pen = (cur == hbfA) ? hbfB : hbfA;
    sage_gemm_bf16<false, 0><<<gemmGrid, dim3(128), 0, stream>>>(
        nullptr, nullptr, cur, nullptr, W1B, b1, pen);
    sage_out_proj<<<(N_NODES + 255) / 256, blk, 0, stream>>>(
        pen, W2, b2, (float*)d_out);
}
